// GraphEncoder_85667417686143
// MI455X (gfx1250) — matrix operand order repaired and checked by reading, not compiled, not run on hardware
//
#include <hip/hip_runtime.h>
#include <hip/hip_bf16.h>
#include <math.h>

typedef __attribute__((ext_vector_type(16))) _Float16 v16h;
typedef __attribute__((ext_vector_type(8)))  _Float16 v8h;
typedef __attribute__((ext_vector_type(8)))  float    v8f;

#define HID 128
#define DIN 64
#define NGRAPH 256
#define LATENT 32

// ---------------------------------------------------------------------------
__global__ void k_fill_f32(float* __restrict__ p, float v, size_t n) {
  size_t i = (size_t)blockIdx.x * blockDim.x + threadIdx.x;
  if (i < n) p[i] = v;
}

// fp32 -> fp16 elementwise
__global__ void k_cvt_f16(const float* __restrict__ src, _Float16* __restrict__ dst, size_t n) {
  size_t i = (size_t)blockIdx.x * blockDim.x + threadIdx.x;
  if (i < n) dst[i] = (_Float16)src[i];
}

// W[K,128] fp32 row-major  ->  Wt[128,K] fp16 (column-major of W)
__global__ void k_wt(const float* __restrict__ W, _Float16* __restrict__ Wt, int K) {
  int idx = blockIdx.x * blockDim.x + threadIdx.x;   // idx = k*128 + n
  if (idx >= K * HID) return;
  int k = idx >> 7, n = idx & 127;
  Wt[(size_t)n * K + k] = (_Float16)W[idx];
}

// degree count: deg[col[e]] += 1  (self loop added later as +1)
__global__ void k_degree(const int* __restrict__ col, float* __restrict__ deg, int E) {
  int e = blockIdx.x * blockDim.x + threadIdx.x;
  if (e < E) atomicAdd(&deg[col[e]], 1.0f);
}

__global__ void k_dinv(float* __restrict__ deg, int n) {
  int i = blockIdx.x * blockDim.x + threadIdx.x;
  if (i < n) deg[i] = rsqrtf(deg[i] + 1.0f);
}

// ---------------------------------------------------------------------------
// WMMA GEMM: C[N,128] = A[N,K](f16) @ W[K,128], W given as Wt[128,K] f16.
// block = 128 thr = 4 waves; block owns 16 rows; wave owns 2 col-tiles of 16.
// A fragment (16x32, ISA layout): lane's halves are two K-contiguous runs of 8
//   -> two 16B loads.  B fragment: same two K-runs of 8 as A -> two 16B loads.
template <int K>
__global__ void k_gemm_f16(const _Float16* __restrict__ A,
                           const _Float16* __restrict__ Wt,
                           float* __restrict__ C) {
  const int lane = threadIdx.x & 31;
  const int wave = threadIdx.x >> 5;
  const int m    = lane & 15;
  const int kgrp = lane >> 4;                 // 0: lanes 0-15, 1: lanes 16-31
  const int rowBase = blockIdx.x * 16;
  const int row = rowBase + m;
  const int nt0 = wave * 2;

  v8f acc0 = {}; v8f acc1 = {};
#pragma unroll
  for (int k0 = 0; k0 < K; k0 += 32) {
    union { v16h v; v8h h[2]; } a;
    const _Float16* ap = A + (size_t)row * K + k0 + (kgrp ? 8 : 0);
    a.h[0] = *(const v8h*)(ap);          // K = base+0..7   -> halves 0..7
    a.h[1] = *(const v8h*)(ap + 16);     // K = base+16..23 -> halves 8..15

#pragma unroll
    for (int t = 0; t < 2; ++t) {
      const int nb = (nt0 + t) * 16 + (lane & 15);
      union { v16h v; v8h h[2]; } b;
      const _Float16* bp = Wt + (size_t)nb * K + k0 + (kgrp ? 8 : 0);
      b.h[0] = *(const v8h*)(bp);        // K = base+0..7
      b.h[1] = *(const v8h*)(bp + 16);   // K = base+16..23
      if (t == 0)
        acc0 = __builtin_amdgcn_wmma_f32_16x16x32_f16(false, a.v, false, b.v, (short)0, acc0, false, false);
      else
        acc1 = __builtin_amdgcn_wmma_f32_16x16x32_f16(false, a.v, false, b.v, (short)0, acc1, false, false);
    }
  }
  // C layout: VGPR v -> row = v + 8*kgrp, col = lane&15
#pragma unroll
  for (int v = 0; v < 8; ++v) {
    int r = rowBase + v + 8 * kgrp;
    C[(size_t)r * HID + (nt0 + 0) * 16 + (lane & 15)] = acc0[v];
    C[(size_t)r * HID + (nt0 + 1) * 16 + (lane & 15)] = acc1[v];
  }
}

// ---------------------------------------------------------------------------
// edge scatter: one wave per edge; lane loads float4 of the source row and
// does 4 contiguous f32 atomics into the target row.
__global__ void k_scatter(const int* __restrict__ row, const int* __restrict__ col,
                          const float* __restrict__ dinv, const float* __restrict__ T,
                          float* __restrict__ agg, int E) {
  int gwid = (int)((blockIdx.x * (size_t)blockDim.x + threadIdx.x) >> 5);
  int lane = threadIdx.x & 31;
  if (gwid >= E) return;
  int j = row[gwid];
  int i = col[gwid];
  float w = dinv[j] * dinv[i];
  const float4 v = ((const float4*)(T + (size_t)j * HID))[lane];
  float* dst = agg + (size_t)i * HID + lane * 4;
  atomicAdd(dst + 0, w * v.x);
  atomicAdd(dst + 1, w * v.y);
  atomicAdd(dst + 2, w * v.z);
  atomicAdd(dst + 3, w * v.w);
}

// post conv1: h = relu(agg + dinv^2 * T + b1) -> f16 buffer for next GEMM
__global__ void k_post1(const float* __restrict__ agg, const float* __restrict__ T,
                        const float* __restrict__ dinv, const float* __restrict__ b1,
                        _Float16* __restrict__ H, size_t total) {
  size_t idx = (size_t)blockIdx.x * blockDim.x + threadIdx.x;
  if (idx >= total) return;
  int i = (int)(idx >> 7);
  int d = (int)(idx & 127);
  float di = dinv[i];
  float v = agg[idx] + di * di * T[idx] + b1[d];
  H[idx] = (_Float16)fmaxf(v, 0.0f);
}

// post conv2: h2 = agg + dinv^2 * T + b2  (in place over agg)
__global__ void k_post2(float* __restrict__ agg, const float* __restrict__ T,
                        const float* __restrict__ dinv, const float* __restrict__ b2,
                        size_t total) {
  size_t idx = (size_t)blockIdx.x * blockDim.x + threadIdx.x;
  if (idx >= total) return;
  int i = (int)(idx >> 7);
  int d = (int)(idx & 127);
  float di = dinv[i];
  agg[idx] = agg[idx] + di * di * T[idx] + b2[d];
}

// gate score: one wave per node, g[i] = dot(h2[i], gate_W) + gate_b
__global__ void k_gate(const float* __restrict__ h2, const float* __restrict__ gW,
                       const float* __restrict__ gb, float* __restrict__ gsc, int n) {
  int gwid = (int)((blockIdx.x * (size_t)blockDim.x + threadIdx.x) >> 5);
  int lane = threadIdx.x & 31;
  if (gwid >= n) return;
  const float4 h = ((const float4*)(h2 + (size_t)gwid * HID))[lane];
  const float4 g = ((const float4*)gW)[lane];
  float s = h.x * g.x + h.y * g.y + h.z * g.z + h.w * g.w;
#pragma unroll
  for (int o = 16; o > 0; o >>= 1) s += __shfl_down(s, o, 32);
  if (lane == 0) gsc[gwid] = s + gb[0];
}

// per-graph segment bounds from sorted batch vector
__global__ void k_bounds_init(int* __restrict__ s, int* __restrict__ e) {
  int t = threadIdx.x;
  if (t < NGRAPH) { s[t] = 0; e[t] = 0; }
}
__global__ void k_bounds(const int* __restrict__ batch, int* __restrict__ s,
                         int* __restrict__ e, int n) {
  int i = blockIdx.x * blockDim.x + threadIdx.x;
  if (i >= n) return;
  int b = batch[i];
  if (i == 0 || batch[i - 1] != b) s[b] = i;
  if (i == n - 1 || batch[i + 1] != b) e[b] = i + 1;
}

// attention pool: one block (128 thr) per graph; softmax over [s,e), weighted sum
__global__ void k_pool(const float* __restrict__ h2, float* __restrict__ gsc,
                       const int* __restrict__ sg, const int* __restrict__ eg,
                       float* __restrict__ pooled) {
  __shared__ float red[HID];
  int g = blockIdx.x, tid = threadIdx.x;
  int s = sg[g], e = eg[g];
  float m = -3.402823466e38f;
  for (int i = s + tid; i < e; i += HID) m = fmaxf(m, gsc[i]);
  red[tid] = m; __syncthreads();
  for (int o = 64; o > 0; o >>= 1) { if (tid < o) red[tid] = fmaxf(red[tid], red[tid + o]); __syncthreads(); }
  float gmax = red[0]; __syncthreads();
  float ssum = 0.0f;
  for (int i = s + tid; i < e; i += HID) { float ev = __expf(gsc[i] - gmax); gsc[i] = ev; ssum += ev; }
  red[tid] = ssum; __syncthreads();
  for (int o = 64; o > 0; o >>= 1) { if (tid < o) red[tid] += red[tid + o]; __syncthreads(); }
  float denom = red[0];
  float inv = denom > 0.0f ? 1.0f / denom : 0.0f;
  float acc = 0.0f;
  for (int i = s; i < e; ++i) acc += gsc[i] * h2[(size_t)i * HID + tid];
  pooled[(size_t)g * HID + tid] = acc * inv;
}

// mu = pooled @ mean_W + mean_b : 256 blocks x 32 threads
__global__ void k_mu(const float* __restrict__ pooled, const float* __restrict__ mW,
                     const float* __restrict__ mb, float* __restrict__ out) {
  int g = blockIdx.x, t = threadIdx.x;
  float acc = mb[t];
#pragma unroll 4
  for (int k = 0; k < HID; ++k) acc += pooled[(size_t)g * HID + k] * mW[(size_t)k * LATENT + t];
  out[(size_t)g * LATENT + t] = acc;
}

// ---------------------------------------------------------------------------
extern "C" void kernel_launch(void* const* d_in, const int* in_sizes, int n_in,
                              void* d_out, int out_size, void* d_ws, size_t ws_size,
                              hipStream_t stream) {
  const float* x     = (const float*)d_in[0];
  const int*   eidx  = (const int*)d_in[1];
  const int*   batch = (const int*)d_in[2];
  const float* W1    = (const float*)d_in[4];
  const float* b1    = (const float*)d_in[5];
  const float* W2    = (const float*)d_in[6];
  const float* b2    = (const float*)d_in[7];
  const float* gateW = (const float*)d_in[8];
  const float* gateB = (const float*)d_in[9];
  const float* meanW = (const float*)d_in[10];
  const float* meanB = (const float*)d_in[11];
  float* out = (float*)d_out;

  const int N = in_sizes[0] / DIN;     // 100000
  const int E = in_sizes[1] / 2;       // 1600000
  const int* erow = eidx;              // sources j
  const int* ecol = eidx + E;          // targets i
  const size_t NH = (size_t)N * HID;
  const size_t ND = (size_t)N * DIN;

  // workspace carve-up (16B-aligned chunks first)
  float*    bufA  = (float*)d_ws;                       // N*128 f32  (xW1, then hW2)
  float*    bufB  = bufA + NH;                          // N*128 f32  (agg1, agg2 -> h2)
  _Float16* bufH  = (_Float16*)(bufB + NH);             // N*128 f16  (relu'd h)
  _Float16* xh    = bufH + NH;                          // N*64  f16  (x converted)
  _Float16* W1t   = xh + ND;                            // 128*64 f16 (W1^T)
  _Float16* W2t   = W1t + (size_t)HID * DIN;            // 128*128 f16 (W2^T)
  float*    dinv  = (float*)(W2t + (size_t)HID * HID);  // N
  float*    gsc   = dinv + N;                           // N
  float*    pooled= gsc + N;                            // 256*128
  int*      sg    = (int*)(pooled + (size_t)NGRAPH * HID);
  int*      eg    = sg + NGRAPH;

  const int mtiles = N / 16;                            // 6250 (exact)
  const int fillNH = (int)((NH + 255) / 256);
  const int scatterBlocks = (E * 32 + 255) / 256;       // one wave per edge

  // 0. precision prep: x -> f16, W1/W2 -> transposed f16
  k_cvt_f16<<<(int)((ND + 255) / 256), 256, 0, stream>>>(x, xh, ND);
  k_wt<<<(DIN * HID + 255) / 256, 256, 0, stream>>>(W1, W1t, DIN);
  k_wt<<<(HID * HID + 255) / 256, 256, 0, stream>>>(W2, W2t, HID);

  // 1. degrees -> dinv
  k_fill_f32<<<(N + 255) / 256, 256, 0, stream>>>(dinv, 0.0f, (size_t)N);
  k_degree<<<(E + 255) / 256, 256, 0, stream>>>(ecol, dinv, E);
  k_dinv<<<(N + 255) / 256, 256, 0, stream>>>(dinv, N);

  // 2. conv1: T1 = x@W1 ; agg1 = scatter ; h = relu(agg1 + dinv^2*T1 + b1)
  k_gemm_f16<DIN><<<mtiles, 128, 0, stream>>>(xh, W1t, bufA);
  k_fill_f32<<<fillNH, 256, 0, stream>>>(bufB, 0.0f, NH);
  k_scatter<<<scatterBlocks, 256, 0, stream>>>(erow, ecol, dinv, bufA, bufB, E);
  k_post1<<<fillNH, 256, 0, stream>>>(bufB, bufA, dinv, b1, bufH, NH);

  // 3. conv2: T2 = h@W2 ; agg2 = scatter ; h2 = agg2 + dinv^2*T2 + b2
  k_fill_f32<<<fillNH, 256, 0, stream>>>(bufB, 0.0f, NH);
  k_gemm_f16<HID><<<mtiles, 128, 0, stream>>>(bufH, W2t, bufA);
  k_scatter<<<scatterBlocks, 256, 0, stream>>>(erow, ecol, dinv, bufA, bufB, E);
  k_post2<<<fillNH, 256, 0, stream>>>(bufB, bufA, dinv, b2, NH);

  // 4. attention pool over sorted batch
  k_gate<<<(N * 32 + 255) / 256, 256, 0, stream>>>(bufB, gateW, gateB, gsc, N);
  k_bounds_init<<<1, 256, 0, stream>>>(sg, eg);
  k_bounds<<<(N + 255) / 256, 256, 0, stream>>>(batch, sg, eg, N);
  k_pool<<<NGRAPH, HID, 0, stream>>>(bufB, gsc, sg, eg, pooled);

  // 5. projection to latent mean
  k_mu<<<NGRAPH, LATENT, 0, stream>>>(pooled, meanW, meanB, out);
}
